// EncoderSRNN_69114613728555
// MI455X (gfx1250) — compile-verified, matrix operand order checked by reading, not run on hardware
//
#include <hip/hip_runtime.h>
#include <hip/hip_bf16.h>

typedef __attribute__((ext_vector_type(16))) __bf16 v16bf;
typedef __attribute__((ext_vector_type(8)))  float  v8f;
typedef unsigned short ushort_t;

#define SEQ 128
#define BATCH 64
#define HID 512
#define ELEM 64
#define SDEPTH 4
#define SSIZE 32
#define NSTACK 2

// ---- fragment helpers --------------------------------------------------
union FragU { uint4 q[2]; v16bf v; };

// A-fragment (16x32 bf16) from an LDS bf16 matrix [16][rowStride]
__device__ inline v16bf load_fragA_lds(const ushort_t* base, int rowStride,
                                       int kt, int lane) {
  int row   = lane & 15;
  int khalf = (lane & 16) ? 8 : 0;
  const ushort_t* p = base + row * rowStride + kt * 32 + khalf;
  FragU f;
  f.q[0] = *(const uint4*)(p);
  f.q[1] = *(const uint4*)(p + 16);
  return f.v;
}

// B-fragment (32x16 bf16) from globally packed fragment store
__device__ inline v16bf load_fragB_g(const uint4* packed, int tileIdx, int lane) {
  const uint4* p = packed + (size_t)(tileIdx * 32 + lane) * 2;
  FragU f;
  f.q[0] = p[0];
  f.q[1] = p[1];
  return f.v;
}

__device__ inline void prefetch_fragB(const uint4* packed, int tileIdx, int lane) {
  __builtin_prefetch(packed + (size_t)(tileIdx * 32 + lane) * 2, 0, 3);
}

__device__ inline v8f wmma_bf16(v16bf a, v16bf b, v8f c) {
  return __builtin_amdgcn_wmma_f32_16x16x32_bf16(false, a, false, b,
                                                 (short)0, c, false, false);
}

__device__ inline ushort_t f2bf(float f) {
  unsigned u = __float_as_uint(f);
  unsigned r = (u + 0x7FFFu + ((u >> 16) & 1u)) >> 16;
  return (ushort_t)r;
}

// ---- workspace layout --------------------------------------------------
// packed B fragments, 1 KB (32 lanes x 32 B) per 16x32 tile:
//   Wi2h : 16x32 tiles  -> 512 tiles, offset     0
//   Wh2h : 16x32 tiles  -> 512 tiles, offset   512
//   Wcat : 16x32 tiles  -> 512 tiles, offset  1024   ([W_s2h^T ; W_s2h^T], K=512)
//   Wh2s : 16x8  tiles  -> 128 tiles, offset  1536
//   Wh2a : 16x1  tiles  ->  16 tiles, offset  1664   (N padded 6 -> 16 with zeros)
#define TILE_WI2H 0
#define TILE_WH2H 512
#define TILE_WCAT 1024
#define TILE_WH2S 1536
#define TILE_WH2A 1664
#define NTILES_TOTAL 1680
#define PACKED_BYTES (NTILES_TOTAL * 1024)

// ------------------------------------------------------------------------
// K0: pack weights into fragment-native bf16 layout
__global__ void srnn_pack_kernel(const float* Wi2h, const float* Wh2h,
                                 const float* Ws2h, const float* Wh2s,
                                 const float* Wh2a, ushort_t* packed) {
  int g = blockIdx.x * blockDim.x + threadIdx.x;
  if (g >= NTILES_TOTAL * 32) return;
  int tile = g >> 5;
  int lane = g & 31;
  int n16  = lane & 15;
  int khalf = (lane & 16) ? 8 : 0;

  ushort_t* out = packed + (size_t)(tile * 32 + lane) * 16;

  int kt, nt;
  const float* W;
  int mat;
  if (tile < TILE_WH2H)      { mat = 0; int t = tile;            kt = t >> 5; nt = t & 31; W = Wi2h; }
  else if (tile < TILE_WCAT) { mat = 1; int t = tile - TILE_WH2H; kt = t >> 5; nt = t & 31; W = Wh2h; }
  else if (tile < TILE_WH2S) { mat = 2; int t = tile - TILE_WCAT; kt = t >> 5; nt = t & 31; W = Ws2h; }
  else if (tile < TILE_WH2A) { mat = 3; int t = tile - TILE_WH2S; kt = t >> 3; nt = t & 7;  W = Wh2s; }
  else                       { mat = 4; int t = tile - TILE_WH2A; kt = t;      nt = 0;      W = Wh2a; }

  int n = nt * 16 + n16;
  for (int j = 0; j < 16; ++j) {
    int koff = khalf + ((j < 8) ? j : (j + 8));
    int k = kt * 32 + koff;
    float v = 0.0f;
    if (mat == 0 || mat == 1)      v = W[n * HID + k];                 // B[k,n] = W[n,k]
    else if (mat == 2)             v = W[n * 256 + (k & 255)];         // concat of two W_s2h^T
    else if (mat == 3)             v = W[n * HID + k];                 // n < 128
    else                           v = (n < 6) ? W[n * HID + k] : 0.f; // zero-padded
    out[j] = f2bf(v);
  }
}

// ------------------------------------------------------------------------
// K1: inp_proj[tb, n] = emb[inputs[tb]] @ W_i2h^T + (b_i2h + b_h2h + 2*b_s2h)
// grid 512 blocks x 128 threads, each block does 16 (t,b) rows.
// Register-blocked: A fragment loaded once per kt, reused for 4 N-tiles.
__global__ void srnn_proj_kernel(const int* inputs, const float* emb,
                                 const ushort_t* packed,
                                 const float* b_i2h, const float* b_h2h,
                                 const float* b_s2h, float* inp_proj) {
  __shared__ ushort_t ab[16 * HID];
  int tid  = threadIdx.x;
  int lane = tid & 31;
  int wave = tid >> 5;
  int tb0  = blockIdx.x * 16;

  for (int idx = tid; idx < 16 * HID; idx += blockDim.x) {
    int i = idx >> 9, k = idx & 511;
    int tok = inputs[tb0 + i];
    ab[idx] = f2bf(emb[(size_t)tok * HID + k]);
  }
  __syncthreads();

  const uint4* pW = (const uint4*)(packed + (size_t)TILE_WI2H * 512);
  for (int g = 0; g < 2; ++g) {
    int ntb = wave * 8 + g * 4;
    v8f acc0 = {}, acc1 = {}, acc2 = {}, acc3 = {};
#pragma unroll 1
    for (int kt = 0; kt < 16; ++kt) {
      if (kt < 15) {
        prefetch_fragB(pW, (kt + 1) * 32 + ntb + 0, lane);
        prefetch_fragB(pW, (kt + 1) * 32 + ntb + 1, lane);
        prefetch_fragB(pW, (kt + 1) * 32 + ntb + 2, lane);
        prefetch_fragB(pW, (kt + 1) * 32 + ntb + 3, lane);
      }
      v16bf a = load_fragA_lds(ab, HID, kt, lane);
      acc0 = wmma_bf16(a, load_fragB_g(pW, kt * 32 + ntb + 0, lane), acc0);
      acc1 = wmma_bf16(a, load_fragB_g(pW, kt * 32 + ntb + 1, lane), acc1);
      acc2 = wmma_bf16(a, load_fragB_g(pW, kt * 32 + ntb + 2, lane), acc2);
      acc3 = wmma_bf16(a, load_fragB_g(pW, kt * 32 + ntb + 3, lane), acc3);
    }
    int n = lane & 15;
#pragma unroll
    for (int it = 0; it < 4; ++it) {
      v8f acc = (it == 0) ? acc0 : (it == 1) ? acc1 : (it == 2) ? acc2 : acc3;
      int col = (ntb + it) * 16 + n;
      float bias = b_i2h[col] + b_h2h[col] + 2.0f * b_s2h[col];
#pragma unroll
      for (int r = 0; r < 8; ++r) {
        int m = r + ((lane & 16) ? 8 : 0);
        inp_proj[(size_t)(tb0 + m) * HID + col] = acc[r] + bias;
      }
    }
  }
}

// ------------------------------------------------------------------------
// K2: persistent recurrence. grid 4 blocks x 512 threads (16 waves),
// each block owns 16 batch rows; all state lives in LDS for 128 steps.
__global__ void __launch_bounds__(512)
srnn_recurrence_kernel(const ushort_t* packed, const float* inp_proj,
                       const float* b_h2a, const float* b_h2s,
                       const float* empty_elem,
                       float* outO, float* outH, float* outS) {
  extern __shared__ char smem[];
  float*    stacksL = (float*)smem;                 // [16][2][32][64]  262144 B
  float*    pvL     = stacksL + 65536;              // [16][128]          8192 B
  float*    logitsL = pvL + 2048;                   // [16][16]           1024 B
  float*    probsL  = logitsL + 256;                // [16][2][3]          384 B
  float*    empL    = probsL + 96;                  // [64]                256 B
  ushort_t* hidL    = (ushort_t*)(empL + 64);       // 2x[16][512] bf16  32768 B
  ushort_t* svL     = hidL + 2 * 16 * HID;          // [16][512]  bf16   16384 B
                                                    // total            321152 B

  int tid  = threadIdx.x;
  int lane = tid & 31;
  int wave = tid >> 5;
  int b0   = blockIdx.x * 16;

  const uint4* pWh2h = (const uint4*)(packed + (size_t)TILE_WH2H * 512);
  const uint4* pWcat = (const uint4*)(packed + (size_t)TILE_WCAT * 512);
  const uint4* pWh2s = (const uint4*)(packed + (size_t)TILE_WH2S * 512);
  const uint4* pWh2a = (const uint4*)(packed + (size_t)TILE_WH2A * 512);

  // init state
  if (tid < 64) empL[tid] = empty_elem[tid];
  for (int idx = tid; idx < 2 * 16 * HID; idx += 512) hidL[idx] = 0;
  __syncthreads();
  for (int idx = tid; idx < 65536; idx += 512) stacksL[idx] = empL[idx & 63];
  __syncthreads();

  int cur = 0;
  for (int t = 0; t < SEQ; ++t) {
    // ---- Phase A: stage stack top-4 as bf16 A-matrix [16][512] ----
    for (int idx = tid; idx < 16 * HID; idx += 512) {
      int m = idx >> 9, k = idx & 511;
      int s = (k >> 8) & 1, d = (k >> 6) & 3, e = k & 63;
      svL[idx] = f2bf(stacksL[m * 4096 + s * 2048 + d * 64 + e]);
    }
    __syncthreads();

    const ushort_t* hcur = hidL + cur * 16 * HID;
    ushort_t*       hnxt = hidL + (cur ^ 1) * 16 * HID;

    // ---- Phase B: all GEMMs read OLD hidden / staged stacks ----
    if (wave < 8) {
      // mid: h @ Wh2h^T + [sv0,sv1] @ [Ws2h^T;Ws2h^T]  (fused K=1024)
      // Register-blocked over 4 N-tiles; A loaded once per kt.
      int ntb = wave * 4;
      v8f acc0 = {}, acc1 = {}, acc2 = {}, acc3 = {};
#pragma unroll 1
      for (int kt = 0; kt < 16; ++kt) {
        if (kt < 15) {
          prefetch_fragB(pWh2h, (kt + 1) * 32 + ntb + 0, lane);
          prefetch_fragB(pWh2h, (kt + 1) * 32 + ntb + 1, lane);
          prefetch_fragB(pWh2h, (kt + 1) * 32 + ntb + 2, lane);
          prefetch_fragB(pWh2h, (kt + 1) * 32 + ntb + 3, lane);
        }
        v16bf a = load_fragA_lds(hcur, HID, kt, lane);
        acc0 = wmma_bf16(a, load_fragB_g(pWh2h, kt * 32 + ntb + 0, lane), acc0);
        acc1 = wmma_bf16(a, load_fragB_g(pWh2h, kt * 32 + ntb + 1, lane), acc1);
        acc2 = wmma_bf16(a, load_fragB_g(pWh2h, kt * 32 + ntb + 2, lane), acc2);
        acc3 = wmma_bf16(a, load_fragB_g(pWh2h, kt * 32 + ntb + 3, lane), acc3);
      }
#pragma unroll 1
      for (int kt = 0; kt < 16; ++kt) {
        if (kt < 15) {
          prefetch_fragB(pWcat, (kt + 1) * 32 + ntb + 0, lane);
          prefetch_fragB(pWcat, (kt + 1) * 32 + ntb + 1, lane);
          prefetch_fragB(pWcat, (kt + 1) * 32 + ntb + 2, lane);
          prefetch_fragB(pWcat, (kt + 1) * 32 + ntb + 3, lane);
        }
        v16bf a = load_fragA_lds(svL, HID, kt, lane);
        acc0 = wmma_bf16(a, load_fragB_g(pWcat, kt * 32 + ntb + 0, lane), acc0);
        acc1 = wmma_bf16(a, load_fragB_g(pWcat, kt * 32 + ntb + 1, lane), acc1);
        acc2 = wmma_bf16(a, load_fragB_g(pWcat, kt * 32 + ntb + 2, lane), acc2);
        acc3 = wmma_bf16(a, load_fragB_g(pWcat, kt * 32 + ntb + 3, lane), acc3);
      }
      int n = lane & 15;
#pragma unroll
      for (int it = 0; it < 4; ++it) {
        v8f acc = (it == 0) ? acc0 : (it == 1) ? acc1 : (it == 2) ? acc2 : acc3;
        int col = (ntb + it) * 16 + n;
#pragma unroll
        for (int r = 0; r < 8; ++r) {
          int m = r + ((lane & 16) ? 8 : 0);
          int gi = (t * BATCH + (b0 + m)) * HID + col;
          float v = tanhf(acc[r] + inp_proj[gi]);
          outO[gi] = v;
          if (t == SEQ - 1) outH[(b0 + m) * HID + col] = v;
          hnxt[m * HID + col] = f2bf(v);
        }
      }
    } else if (wave < 10) {
      // push_vals = tanh(h @ Wh2s^T + b_h2s); register-blocked 4 N-tiles
      int ntb = (wave - 8) * 4;
      v8f acc0 = {}, acc1 = {}, acc2 = {}, acc3 = {};
#pragma unroll 1
      for (int kt = 0; kt < 16; ++kt) {
        if (kt < 15) {
          prefetch_fragB(pWh2s, (kt + 1) * 8 + ntb + 0, lane);
          prefetch_fragB(pWh2s, (kt + 1) * 8 + ntb + 1, lane);
          prefetch_fragB(pWh2s, (kt + 1) * 8 + ntb + 2, lane);
          prefetch_fragB(pWh2s, (kt + 1) * 8 + ntb + 3, lane);
        }
        v16bf a = load_fragA_lds(hcur, HID, kt, lane);
        acc0 = wmma_bf16(a, load_fragB_g(pWh2s, kt * 8 + ntb + 0, lane), acc0);
        acc1 = wmma_bf16(a, load_fragB_g(pWh2s, kt * 8 + ntb + 1, lane), acc1);
        acc2 = wmma_bf16(a, load_fragB_g(pWh2s, kt * 8 + ntb + 2, lane), acc2);
        acc3 = wmma_bf16(a, load_fragB_g(pWh2s, kt * 8 + ntb + 3, lane), acc3);
      }
      int n = lane & 15;
#pragma unroll
      for (int it = 0; it < 4; ++it) {
        v8f acc = (it == 0) ? acc0 : (it == 1) ? acc1 : (it == 2) ? acc2 : acc3;
        int col = (ntb + it) * 16 + n;
        float bias = b_h2s[col];
#pragma unroll
        for (int r = 0; r < 8; ++r) {
          int m = r + ((lane & 16) ? 8 : 0);
          pvL[m * 128 + col] = tanhf(acc[r] + bias);
        }
      }
    } else if (wave == 10) {
      // action logits: h @ Wh2a^T (N padded to 16)
      v8f acc = {};
#pragma unroll 1
      for (int kt = 0; kt < 16; ++kt) {
        v16bf a = load_fragA_lds(hcur, HID, kt, lane);
        v16bf b = load_fragB_g(pWh2a, kt, lane);
        acc = wmma_bf16(a, b, acc);
      }
      int n = lane & 15;
#pragma unroll
      for (int r = 0; r < 8; ++r) {
        int m = r + ((lane & 16) ? 8 : 0);
        logitsL[m * 16 + n] = acc[r];
      }
    }
    __syncthreads();

    // ---- Phase C: softmax over 3 actions per (row, stack) ----
    if (tid < 32) {
      int m = tid >> 1, s = tid & 1;
      float l0 = logitsL[m * 16 + s * 3 + 0] + b_h2a[s * 3 + 0];
      float l1 = logitsL[m * 16 + s * 3 + 1] + b_h2a[s * 3 + 1];
      float l2 = logitsL[m * 16 + s * 3 + 2] + b_h2a[s * 3 + 2];
      float mx = fmaxf(fmaxf(l0, l1), l2);
      float e0 = __expf(l0 - mx), e1 = __expf(l1 - mx), e2 = __expf(l2 - mx);
      float inv = 1.0f / (e0 + e1 + e2);
      probsL[(m * 2 + s) * 3 + 0] = e0 * inv;
      probsL[(m * 2 + s) * 3 + 1] = e1 * inv;
      probsL[(m * 2 + s) * 3 + 2] = e2 * inv;
    }
    __syncthreads();

    // ---- Phase D: 3-tap stack blend (rolling, in place) ----
    for (int c = tid; c < 2048; c += 512) {
      int m = c >> 7, s = (c >> 6) & 1, e = c & 63;
      const float* pr = &probsL[(m * 2 + s) * 3];
      float pp = pr[0], po = pr[1], pn = pr[2];
      float pv = pvL[m * 128 + s * 64 + e];
      float* col = &stacksL[(m * 2 + s) * 2048 + e];
      float prev = col[0];
      float curv = col[64];
      col[0] = pp * pv;
#pragma unroll 1
      for (int d = 1; d < 31; ++d) {
        float nxtv = col[(d + 1) * 64];
        col[d * 64] = pp * prev + po * nxtv + pn * curv;
        prev = curv;
        curv = nxtv;
      }
      col[31 * 64] = empL[e];
    }
    __syncthreads();
    cur ^= 1;
  }

  // final stacks -> global
  for (int idx = tid; idx < 65536; idx += 512) {
    int m = idx >> 12, rem = idx & 4095;
    outS[(b0 + m) * 4096 + rem] = stacksL[idx];
  }
}

// ------------------------------------------------------------------------
extern "C" void kernel_launch(void* const* d_in, const int* in_sizes, int n_in,
                              void* d_out, int out_size, void* d_ws, size_t ws_size,
                              hipStream_t stream) {
  const int*   inputs = (const int*)  d_in[0];
  const float* emb    = (const float*)d_in[1];
  const float* W_i2h  = (const float*)d_in[2];
  const float* b_i2h  = (const float*)d_in[3];
  const float* W_h2h  = (const float*)d_in[4];
  const float* b_h2h  = (const float*)d_in[5];
  const float* W_h2a  = (const float*)d_in[6];
  const float* b_h2a  = (const float*)d_in[7];
  const float* W_h2s  = (const float*)d_in[8];
  const float* b_h2s  = (const float*)d_in[9];
  const float* W_s2h  = (const float*)d_in[10];
  const float* b_s2h  = (const float*)d_in[11];
  const float* empty_elem = (const float*)d_in[12];

  ushort_t* packed   = (ushort_t*)d_ws;
  float*    inp_proj = (float*)((char*)d_ws + PACKED_BYTES);

  float* outO = (float*)d_out;                          // (128,64,512)
  float* outH = outO + (size_t)SEQ * BATCH * HID;       // (64,512)
  float* outS = outH + (size_t)BATCH * HID;             // (64,2,32,64)

  srnn_pack_kernel<<<(NTILES_TOTAL * 32 + 255) / 256, 256, 0, stream>>>(
      W_i2h, W_h2h, W_s2h, W_h2s, W_h2a, packed);

  srnn_proj_kernel<<<(SEQ * BATCH) / 16, 128, 0, stream>>>(
      inputs, emb, packed, b_i2h, b_h2h, b_s2h, inp_proj);

  srnn_recurrence_kernel<<<BATCH / 16, 512, 321152, stream>>>(
      packed, inp_proj, b_h2a, b_h2s, empty_elem, outO, outH, outS);
}
